// S4Base_46918222742378
// MI455X (gfx1250) — matrix operand order repaired and checked by reading, not compiled, not run on hardware
//
#include <hip/hip_runtime.h>

// S4 forward: y = causal_conv(u, K) + D*u.
//  Kernel A : per-channel spectrum evaluation + 16K-pt inverse FFT in LDS -> K (f16, padded)
//  Kernel A2: materialize Toeplitz A-fragments (per-lane WMMA layout) into global scratch
//  Kernel B : implicit-Toeplitz GEMM, A-fragments staged per-iteration by the
//             Tensor Data Mover (16KB tensor_load_to_lds), v_wmma_f32_16x16x32_f16
//  Kernel Bf: fallback (small scratch): A built by scalar gather from a TDM-staged K window

#define SEQ_L 16384
#define NSTATE 64
#define NCHAN 64
#define NBATCH 16
#define KPAD 320
#define KROW (SEQ_L + 2 * KPAD)
#define NDIAG 1024                 // distinct 16-row Toeplitz diagonals per channel

typedef _Float16 v16h __attribute__((ext_vector_type(16)));
typedef float v8f __attribute__((ext_vector_type(8)));
typedef unsigned int u32x4 __attribute__((ext_vector_type(4)));
typedef int i32x4 __attribute__((ext_vector_type(4)));
typedef int i32x8 __attribute__((ext_vector_type(8)));

struct cf { float x, y; };
__device__ __forceinline__ cf cmul(cf a, cf b) {
    return { a.x * b.x - a.y * b.y, a.x * b.y + a.y * b.x };
}
__device__ __forceinline__ cf cadd(cf a, cf b) { return { a.x + b.x, a.y + b.y }; }
__device__ __forceinline__ cf csub(cf a, cf b) { return { a.x - b.x, a.y - b.y }; }
__device__ __forceinline__ cf cinv(cf a) {
    float r = 1.0f / (a.x * a.x + a.y * a.y);
    return { a.x * r, -a.y * r };
}
__device__ __forceinline__ cf cdiv(cf a, cf b) { return cmul(a, cinv(b)); }

// A-matrix k order chosen here (contiguous, not the ISA table order; B uses the same): lane = (m = lane&15, h = lane>>4),
// vector element e holds K-index kA(e,h) = e + 16h.
__device__ __forceinline__ int kA(int e, int h) { return e + 16 * h; }

// ---------------------------------------------------------------------------
// Kernel A: one workgroup per channel d. Evaluate generating function at the
// L roots of unity (stored bit-reversed into LDS), in-place inverse DIT FFT,
// write real part as f16 into a zero-padded per-channel row of Kh.
// ---------------------------------------------------------------------------
__global__ __launch_bounds__(1024) void s4_eval_ifft_kernel(
    const float* __restrict__ Lam_ri, const float* __restrict__ P_ri,
    const float* __restrict__ B_ri, const float* __restrict__ C_ri,
    const float* __restrict__ log_step, _Float16* __restrict__ Kh)
{
    __shared__ cf F[SEQ_L];                 // 128 KB (fits 320 KB WGP LDS)
    __shared__ cf lamS[NSTATE], a0S[NSTATE], pS[NSTATE], bS[NSTATE];

    const int d = blockIdx.x;
    const int tid = threadIdx.x;

    if (tid < NSTATE) {
        lamS[tid] = { Lam_ri[2 * tid], Lam_ri[2 * tid + 1] };
        pS[tid]   = { P_ri[2 * tid],   P_ri[2 * tid + 1] };
        bS[tid]   = { B_ri[2 * tid],   B_ri[2 * tid + 1] };
        a0S[tid]  = { C_ri[2 * (d * NSTATE + tid)], -C_ri[2 * (d * NSTATE + tid) + 1] };
    }
    __syncthreads();

    const float step = __expf(log_step[0]);
    const float two_over_step = 2.0f / step;
    const float TWO_PI = 6.28318530717958647692f;

    for (int l = tid; l < SEQ_L; l += 1024) {
        float sn, cs;
        __sincosf(TWO_PI * ((float)l / (float)SEQ_L), &sn, &cs);
        cf Om    = { cs, -sn };                       // exp(-2*pi*i*l/L)
        cf onepO = { 1.0f + Om.x, Om.y };
        cf onemO = { 1.0f - Om.x, -Om.y };
        cf q = cdiv(onemO, onepO);
        cf g = { two_over_step * q.x, two_over_step * q.y };

        cf s00 = {0, 0}, s01 = {0, 0}, s10 = {0, 0}, s11 = {0, 0};
        for (int n = 0; n < NSTATE; n++) {
            cf den = { g.x - lamS[n].x, g.y - lamS[n].y };
            cf cau = cinv(den);
            cf p  = pS[n];
            cf a1 = { p.x, -p.y };                    // conj(P)
            cf a0c = cmul(a0S[n], cau);
            cf a1c = cmul(a1, cau);
            s00 = cadd(s00, cmul(a0c, bS[n]));
            s01 = cadd(s01, cmul(a0c, p));
            s10 = cadd(s10, cmul(a1c, bS[n]));
            s11 = cadd(s11, cmul(a1c, p));
        }
        cf inner = csub(s00, cmul(s01, cmul(s10, cinv({ 1.0f + s11.x, s11.y }))));
        cf two = { 2.0f, 0.0f };
        cf ev = cmul(cdiv(two, onepO), inner);

        unsigned r = __brev((unsigned)l) >> 18;       // 14-bit bit reversal
        F[r] = ev;                                    // bit-reversed store
    }
    __syncthreads();

    for (int s = 1; s <= 14; s++) {
        const int half = 1 << (s - 1);
        const int len  = 1 << s;
        for (int i = tid; i < (SEQ_L >> 1); i += 1024) {
            const int blk = i >> (s - 1);
            const int j   = i & (half - 1);
            const int pos = blk * len + j;
            float sn, cs;
            __sincosf(TWO_PI * (float)j / (float)len, &sn, &cs);
            cf w = { cs, sn };                        // exp(+2*pi*i*j/len)
            cf a = F[pos];
            cf t = cmul(F[pos + half], w);
            F[pos]        = { a.x + t.x, a.y + t.y };
            F[pos + half] = { a.x - t.x, a.y - t.y };
        }
        __syncthreads();
    }

    _Float16* row = Kh + (size_t)d * KROW;
    for (int i = tid; i < KPAD; i += 1024) {
        row[i] = (_Float16)0.0f;
        row[KPAD + SEQ_L + i] = (_Float16)0.0f;
    }
    const float scale = 1.0f / (float)SEQ_L;
    for (int t = tid; t < SEQ_L; t += 1024)
        row[KPAD + t] = (_Float16)(F[t].x * scale);
}

// ---------------------------------------------------------------------------
// Kernel A2: materialize all Toeplitz A-fragments in WMMA per-lane layout.
// AF[((d*NDIAG + d16)*32 + lane)] : v16h with element e = K[d16*16 + m - kA(e,h)].
// One thread per (d, d16, lane).  grid = (NCHAN, NDIAG*32/256).
// ---------------------------------------------------------------------------
__global__ __launch_bounds__(256) void s4_afrag_build_kernel(
    const _Float16* __restrict__ Kh, v16h* __restrict__ AF)
{
    const int d = blockIdx.x;
    const int t = blockIdx.y * 256 + threadIdx.x;    // 0 .. NDIAG*32-1
    const int d16 = t >> 5;
    const int lane = t & 31;
    const int m = lane & 15;
    const int h = lane >> 4;

    const _Float16* row = Kh + (size_t)d * KROW + KPAD;  // row[x] = K[x], pad-safe
    v16h v;
    #pragma unroll
    for (int e = 0; e < 16; e++) {
        const int x = d16 * 16 + m - kA(e, h);       // x in [-31, SEQ_L-1]
        v[e] = row[x];
    }
    AF[((size_t)d * NDIAG + (size_t)d16) * 32 + lane] = v;
}

// ---------------------------------------------------------------------------
// Kernel B (optimized): grid = (d, t-macro of 256 rows), block = 256 (8 waves).
// Per 32-wide s-block: TDM loads the 16 needed diagonal A-tiles (16KB) to LDS,
// threads stage u as f16, each wave does two aligned v16h LDS reads for A,
// one for B, and two WMMAs.
// ---------------------------------------------------------------------------
__global__ __launch_bounds__(256) void s4_conv_wmma_af_kernel(
    const float* __restrict__ u, const v16h* __restrict__ AF,
    const float* __restrict__ Dp, float* __restrict__ out)
{
    __shared__ __align__(1024) v16h AFlds[16 * 32];     // 16 tiles x 1KB = 16KB
    __shared__ __align__(32) _Float16 uH[NBATCH * 32];  // [b][s] f16

    const int d   = blockIdx.x;
    const int t0m = blockIdx.y * 256;
    const int tid  = threadIdx.x;
    const int wave = tid >> 5;
    const int lane = tid & 31;
    const int m = lane & 15;
    const int h = lane >> 4;
    const int t0 = t0m + wave * 32;                  // tiles [t0,t0+15], [t0+16,t0+31]

    v8f c0 = {};
    v8f c1 = {};

    const unsigned af_lds = (unsigned)(size_t)(&AFlds[0]);

    for (int s0 = 0; s0 < t0m + 256; s0 += 32) {
        const int dbase  = (t0m - s0) >> 4;          // may be negative (exact /16)
        const int dstart = (dbase > 0) ? dbase : 0;  // clamp to valid diagonals

        // --- TDM: 16 contiguous diagonal tiles (16384 halves) -> LDS ------
        if (wave == 0) {
            unsigned long long gaddr = (unsigned long long)(const void*)
                (AF + ((size_t)d * NDIAG + (size_t)dstart) * 32);
            u32x4 g0 = { 1u,                               // count=1, valid
                         af_lds,                           // lds_addr
                         (unsigned)gaddr,
                         (unsigned)(gaddr >> 32) | (2u << 30) }; // type=2 (image)
            i32x8 g1 = { (int)(1u << 16),     // data_size=1 (2 bytes)
                         (int)(16384u << 16), // tensor_dim0 = 16384 (bits 79:48)
                         (int)(1u << 16),     // tensor_dim1 = 1     (bits 111:80)
                         (int)(16384u << 16), // tile_dim0   = 16384 (bits 127:112)
                         (int)(1u | (1u << 16)), // tile_dim1=1, tile_dim2=1
                         16384,               // tensor_dim0_stride low
                         0, 0 };
            i32x4 g2 = { 1, 0, 0, 0 };        // tensor_dim2 = 1
            i32x4 g3 = { 0, 0, 0, 0 };
            i32x8 g4 = { 0, 0, 0, 0, 0, 0, 0, 0 };
            __builtin_amdgcn_tensor_load_to_lds(g0, g1, g2, g3, g4, 0);
        }

        // --- stage u block: uH[b][s] = (f16) u[b, s0+s, d] ----------------
        #pragma unroll
        for (int k = 0; k < 2; k++) {
            const int f = tid + k * 256;
            const int b = f >> 5;
            const int s = f & 31;
            uH[b * 32 + s] =
                (_Float16)u[((size_t)b * SEQ_L + (size_t)(s0 + s)) * NCHAN + d];
        }

        if (wave == 0) __builtin_amdgcn_s_wait_tensorcnt(0);
        __syncthreads();

        // --- compute -------------------------------------------------------
        if (s0 <= t0 + 31) {
            // B fragment: element e = u[b=m, s0 + e + 16h, d]
            v16h bfrag = *(const v16h*)(&uH[m * 32 + 16 * h]);

            {   // tile rows t0+16 .. t0+31
                const int idx1 = ((t0 + 16 - s0) >> 4) - dstart;
                v16h a1f = AFlds[idx1 * 32 + lane];
                c1 = __builtin_amdgcn_wmma_f32_16x16x32_f16(
                    false, a1f, false, bfrag, (short)0, c1, false, false);
            }
            if (s0 <= t0 + 15) {   // tile rows t0 .. t0+15
                const int idx0 = ((t0 - s0) >> 4) - dstart;
                v16h a0f = AFlds[idx0 * 32 + lane];
                c0 = __builtin_amdgcn_wmma_f32_16x16x32_f16(
                    false, a0f, false, bfrag, (short)0, c0, false, false);
            }
        }
        __syncthreads();
    }

    const float dcoef = Dp[d];
    const int b = lane & 15;
    #pragma unroll
    for (int r = 0; r < 8; r++) {
        const int mm = r + 8 * h;
        {
            const size_t idx = ((size_t)b * SEQ_L + (size_t)(t0 + mm)) * NCHAN + d;
            out[idx] = c0[r] + dcoef * u[idx];
        }
        {
            const size_t idx = ((size_t)b * SEQ_L + (size_t)(t0 + 16 + mm)) * NCHAN + d;
            out[idx] = c1[r] + dcoef * u[idx];
        }
    }
}

// ---------------------------------------------------------------------------
// Kernel Bf (fallback, small scratch): TDM stages a 288-half K window; A-frags
// gathered by scalar LDS reads.  Same math, same output.
// ---------------------------------------------------------------------------
__global__ __launch_bounds__(256) void s4_conv_wmma_win_kernel(
    const float* __restrict__ u, const _Float16* __restrict__ Kh,
    const float* __restrict__ Dp, float* __restrict__ out)
{
    __shared__ __align__(16) _Float16 KwinS[320];
    __shared__ __align__(32) _Float16 uH[NBATCH * 32];

    const int d   = blockIdx.x;
    const int t0m = blockIdx.y * 256;
    const int tid  = threadIdx.x;
    const int wave = tid >> 5;
    const int lane = tid & 31;
    const int m = lane & 15;
    const int h = lane >> 4;
    const int t0 = t0m + wave * 32;

    v8f c0 = {};
    v8f c1 = {};

    const unsigned kwin_lds = (unsigned)(size_t)(&KwinS[0]);

    for (int s0 = 0; s0 < t0m + 256; s0 += 32) {
        if (wave == 0) {
            unsigned long long gaddr = (unsigned long long)(const void*)
                (Kh + (size_t)d * KROW + (KPAD + t0m - s0 - 31));
            u32x4 g0 = { 1u, kwin_lds, (unsigned)gaddr,
                         (unsigned)(gaddr >> 32) | (2u << 30) };
            i32x8 g1 = { (int)(1u << 16), (int)(288u << 16), (int)(1u << 16),
                         (int)(288u << 16), (int)(1u | (1u << 16)), 288, 0, 0 };
            i32x4 g2 = { 1, 0, 0, 0 };
            i32x4 g3 = { 0, 0, 0, 0 };
            i32x8 g4 = { 0, 0, 0, 0, 0, 0, 0, 0 };
            __builtin_amdgcn_tensor_load_to_lds(g0, g1, g2, g3, g4, 0);
        }

        #pragma unroll
        for (int k = 0; k < 2; k++) {
            const int f = tid + k * 256;
            const int b = f >> 5;
            const int s = f & 31;
            uH[b * 32 + s] =
                (_Float16)u[((size_t)b * SEQ_L + (size_t)(s0 + s)) * NCHAN + d];
        }

        if (wave == 0) __builtin_amdgcn_s_wait_tensorcnt(0);
        __syncthreads();

        if (s0 <= t0 + 31) {
            v16h bfrag = *(const v16h*)(&uH[m * 32 + 16 * h]);
            {
                v16h a1f;
                #pragma unroll
                for (int e = 0; e < 16; e++)
                    a1f[e] = KwinS[wave * 32 + 16 + m + 31 - kA(e, h)];
                c1 = __builtin_amdgcn_wmma_f32_16x16x32_f16(
                    false, a1f, false, bfrag, (short)0, c1, false, false);
            }
            if (s0 <= t0 + 15) {
                v16h a0f;
                #pragma unroll
                for (int e = 0; e < 16; e++)
                    a0f[e] = KwinS[wave * 32 + m + 31 - kA(e, h)];
                c0 = __builtin_amdgcn_wmma_f32_16x16x32_f16(
                    false, a0f, false, bfrag, (short)0, c0, false, false);
            }
        }
        __syncthreads();
    }

    const float dcoef = Dp[d];
    const int b = lane & 15;
    #pragma unroll
    for (int r = 0; r < 8; r++) {
        const int mm = r + 8 * h;
        {
            const size_t idx = ((size_t)b * SEQ_L + (size_t)(t0 + mm)) * NCHAN + d;
            out[idx] = c0[r] + dcoef * u[idx];
        }
        {
            const size_t idx = ((size_t)b * SEQ_L + (size_t)(t0 + 16 + mm)) * NCHAN + d;
            out[idx] = c1[r] + dcoef * u[idx];
        }
    }
}

// ---------------------------------------------------------------------------
extern "C" void kernel_launch(void* const* d_in, const int* in_sizes, int n_in,
                              void* d_out, int out_size, void* d_ws, size_t ws_size,
                              hipStream_t stream) {
    (void)in_sizes; (void)n_in; (void)out_size;
    const float* u      = (const float*)d_in[0];
    const float* Lam_ri = (const float*)d_in[1];
    const float* P_ri   = (const float*)d_in[2];
    const float* B_ri   = (const float*)d_in[3];
    const float* C_ri   = (const float*)d_in[4];
    const float* Dp     = (const float*)d_in[5];
    const float* lstep  = (const float*)d_in[6];

    const size_t AF_BYTES = (size_t)NCHAN * NDIAG * 32 * sizeof(v16h); // 64 MB
    const size_t KH_BYTES = (size_t)NCHAN * KROW * sizeof(_Float16);   // ~2.2 MB
    const bool big_ws = (ws_size >= AF_BYTES + KH_BYTES);

    dim3 gridB(NCHAN, SEQ_L / 256);

    if (big_ws) {
        v16h*     AF = (v16h*)d_ws;
        _Float16* Kh = (_Float16*)((char*)d_ws + AF_BYTES);
        s4_eval_ifft_kernel<<<NCHAN, 1024, 0, stream>>>(Lam_ri, P_ri, B_ri, C_ri,
                                                        lstep, Kh);
        dim3 gridA2(NCHAN, NDIAG * 32 / 256);
        s4_afrag_build_kernel<<<gridA2, 256, 0, stream>>>(Kh, AF);
        s4_conv_wmma_af_kernel<<<gridB, 256, 0, stream>>>(u, AF, Dp, (float*)d_out);
    } else {
        _Float16* Kh = (_Float16*)d_ws;
        s4_eval_ifft_kernel<<<NCHAN, 1024, 0, stream>>>(Lam_ri, P_ri, B_ri, C_ri,
                                                        lstep, Kh);
        s4_conv_wmma_win_kernel<<<gridB, 256, 0, stream>>>(u, Kh, Dp, (float*)d_out);
    }
}
